// CasualSelfAttention_88579405513078
// MI455X (gfx1250) — compile-verified, matrix operand order checked by reading, not run on hardware
//
#include <hip/hip_runtime.h>

// Problem constants (match reference)
#define B_ 2
#define T_ 4096
#define C_ 768
#define H_ 12
#define D_ 64
#define M_ (B_ * T_)   // 8192 total tokens

typedef __attribute__((ext_vector_type(16))) _Float16 v16h;
typedef __attribute__((ext_vector_type(8)))  _Float16 v8h;
typedef __attribute__((ext_vector_type(8)))  float    v8f;

// ---------------------------------------------------------------------------
// Fragment loaders.
// CDNA5 16-bit A-matrix 16x32 layout (wave32): lane L (0..15) holds row M=L,
// halves 0..7 = K kb..kb+7, halves 8..15 = K kb+16..kb+23 with kb=k0;
// lanes 16..31 hold same rows with kb=k0+8 (K 8..15 / 24..31).  B (32x16) is
// the mirror with the lane holding column N.  So a fragment is always two
// contiguous 16-byte runs at [base] and [base+16 halves].
// ---------------------------------------------------------------------------
__device__ inline v16h frag_from_f32(const float* __restrict__ p) {
  v16h f;
#pragma unroll
  for (int i = 0; i < 8; ++i) f[i] = (_Float16)p[i];
#pragma unroll
  for (int i = 0; i < 8; ++i) f[8 + i] = (_Float16)p[16 + i];
  return f;
}

__device__ inline v16h frag_from_f16(const _Float16* p) {
  v8h lo = *(const v8h*)(p);
  v8h hi = *(const v8h*)(p + 16);
  v16h f;
#pragma unroll
  for (int i = 0; i < 8; ++i) { f[i] = lo[i]; f[8 + i] = hi[i]; }
  return f;
}

__device__ inline v8f wmma16(v16h a, v16h b, v8f c) {
  // v_wmma_f32_16x16x32_f16 : D = A(16x32) * B(32x16) + C, fp32 accumulate
  return __builtin_amdgcn_wmma_f32_16x16x32_f16(false, a, false, b, (short)0, c,
                                                false, false);
}

// ---------------------------------------------------------------------------
// Projection GEMM: Y[M x 768] = A[M x 768] @ W[768 x 768]^T + bias, epilogue
// scale, flexible output layout.
//   OUT_MODE 0 : f16, head-major [B*H][T][D]            (Q, K)
//   OUT_MODE 1 : f16, head-major transposed [B*H][D][T] (V^T)
//   OUT_MODE 2 : f32, row-major [M][768]                (final output)
// Block = 128 threads (4 waves); each wave computes a 32(M) x 64(N) tile:
// 2 A frags + 4 B frags -> 8 WMMAs per k-step of 32.  Weights (2.3 MB) are
// fully L2-resident, so redundant B-fragment traffic is cheap.
// ---------------------------------------------------------------------------
template <int OUT_MODE, bool A_IS_F16>
__global__ __launch_bounds__(128)
void proj_gemm(const void* __restrict__ Aptr, const float* __restrict__ W,
               const float* __restrict__ bias, void* __restrict__ outp,
               float scale) {
  constexpr int K = C_;
  const int lane = threadIdx.x & 31;
  const int wave = threadIdx.x >> 5;
  const int m0 = (blockIdx.x * 4 + wave) * 32;
  const int n0 = blockIdx.y * 64;
  const int hs = (lane >> 4) * 8;   // half-wave K offset (0 or 8)
  const int rA = lane & 15;         // row (A) / column (B,C)

  v8f acc[2][4] = {};

  for (int k0 = 0; k0 < K; k0 += 32) {
    const int kb = k0 + hs;
    v16h a0, a1;
    if (A_IS_F16) {
      const _Float16* A = (const _Float16*)Aptr;
      a0 = frag_from_f16(A + (size_t)(m0 + rA) * K + kb);
      a1 = frag_from_f16(A + (size_t)(m0 + 16 + rA) * K + kb);
    } else {
      const float* A = (const float*)Aptr;
      a0 = frag_from_f32(A + (size_t)(m0 + rA) * K + kb);
      a1 = frag_from_f32(A + (size_t)(m0 + 16 + rA) * K + kb);
    }
#pragma unroll
    for (int t = 0; t < 4; ++t) {
      // B[k][n] = W[n][k]; lane holds column n -> contiguous row of W
      v16h bf = frag_from_f32(W + (size_t)(n0 + t * 16 + rA) * K + kb);
      acc[0][t] = wmma16(a0, bf, acc[0][t]);
      acc[1][t] = wmma16(a1, bf, acc[1][t]);
    }
  }

  // C/D layout: lane<16 -> n=lane, rows r=0..7 ; lane>=16 -> n=lane-16, rows 8+r
#pragma unroll
  for (int i = 0; i < 2; ++i) {
#pragma unroll
    for (int t = 0; t < 4; ++t) {
      const int n = n0 + t * 16 + rA;
      const float bv = bias[n];
      const int mb = m0 + i * 16 + hs;
#pragma unroll
      for (int r = 0; r < 8; ++r) {
        const int m = mb + r;
        const float v = (acc[i][t][r] + bv) * scale;
        if (OUT_MODE == 2) {
          ((float*)outp)[(size_t)m * C_ + n] = v;
        } else {
          const int b = m >> 12, tok = m & (T_ - 1);
          const int h = n >> 6, d = n & (D_ - 1);
          const size_t idx =
              (OUT_MODE == 0)
                  ? (((size_t)(b * H_ + h) * T_ + tok) * D_ + d)
                  : (((size_t)(b * H_ + h) * D_ + d) * T_ + tok);
          ((_Float16*)outp)[idx] = (_Float16)v;
        }
      }
    }
  }
}

// ---------------------------------------------------------------------------
// Flash attention (causal, online softmax). Grid (T/64, B*H); 4 waves/block,
// each wave owns a 16-query tile (scores never touch global memory).
// 64-key blocks: 8 WMMAs (S = Q K^T, 4 tiles x 2 K-steps), softmax in
// registers with 16-lane butterflies (amortized over 64 keys), P staged
// through per-wave LDS (C-layout -> A-layout), 8 WMMAs (O += P V).
// Q pre-scaled by 1/sqrt(D) at projection time.
// ---------------------------------------------------------------------------
__global__ __launch_bounds__(128)
void flash_attn(const _Float16* __restrict__ Q, const _Float16* __restrict__ Kh,
                const _Float16* __restrict__ Vt, _Float16* __restrict__ Out) {
  __shared__ __align__(16) _Float16 plds[4][16 * 64];
  const int lane = threadIdx.x & 31;
  const int wave = threadIdx.x >> 5;
  const int bh = blockIdx.y;
  const int b = bh / H_;
  const int h = bh - b * H_;
  const int q0 = blockIdx.x * 64 + wave * 16;
  const int hs = (lane >> 4) * 8;
  const int rA = lane & 15;

  const _Float16* Qp = Q + (size_t)bh * T_ * D_;
  const _Float16* Kp = Kh + (size_t)bh * T_ * D_;
  const _Float16* Vp = Vt + (size_t)bh * D_ * T_;
  _Float16* P = plds[wave];

  // Q tile as A-fragments over D=64 (two K-steps of 32)
  v16h qf0 = frag_from_f16(Qp + (size_t)(q0 + rA) * D_ + hs);
  v16h qf1 = frag_from_f16(Qp + (size_t)(q0 + rA) * D_ + 32 + hs);

  v8f o[4] = {};              // 16 x 64 output accumulator (4 C-fragments)
  float rmax[8], rsum[8];     // per-row stats, replicated across 16-lane half
#pragma unroll
  for (int r = 0; r < 8; ++r) { rmax[r] = -__builtin_inff(); rsum[r] = 0.0f; }

  const int qrow = q0 + hs;   // this lane's rows are qrow+0 .. qrow+7

  for (int j0 = 0; j0 <= q0 + 15; j0 += 64) {
    // ---- S = Q K^T for 64 keys (four 16x16 tiles) ----
    v8f s[4] = {};
#pragma unroll
    for (int t = 0; t < 4; ++t) {
      v16h kf0 = frag_from_f16(Kp + (size_t)(j0 + t * 16 + rA) * D_ + hs);
      v16h kf1 = frag_from_f16(Kp + (size_t)(j0 + t * 16 + rA) * D_ + 32 + hs);
      s[t] = wmma16(qf0, kf0, s[t]);
      s[t] = wmma16(qf1, kf1, s[t]);
    }
    // ---- causal mask (only partial blocks; wave-uniform branch) ----
    if (j0 + 63 > q0) {
      const float NEG = -__builtin_inff();
#pragma unroll
      for (int t = 0; t < 4; ++t)
#pragma unroll
        for (int r = 0; r < 8; ++r)
          s[t][r] = (j0 + t * 16 + rA <= qrow + r) ? s[t][r] : NEG;
    }
    // ---- online softmax: per-row max/sum via 16-lane butterflies ----
    float mnew[8], sc[8], bs[8];
#pragma unroll
    for (int r = 0; r < 8; ++r) {
      float bm = fmaxf(fmaxf(s[0][r], s[1][r]), fmaxf(s[2][r], s[3][r]));
#pragma unroll
      for (int off = 1; off < 16; off <<= 1)
        bm = fmaxf(bm, __shfl_xor(bm, off, 32));
      mnew[r] = fmaxf(rmax[r], bm);
      sc[r] = __expf(rmax[r] - mnew[r]);
      rmax[r] = mnew[r];
    }
#pragma unroll
    for (int r = 0; r < 8; ++r) {
      float ssum = 0.0f;
#pragma unroll
      for (int t = 0; t < 4; ++t) {
        float p = __expf(s[t][r] - mnew[r]);
        // stage P (16x64 f16) in per-wave LDS, C-layout addressing
        P[(hs + r) * 64 + t * 16 + rA] = (_Float16)p;
        ssum += p;
      }
#pragma unroll
      for (int off = 1; off < 16; off <<= 1)
        ssum += __shfl_xor(ssum, off, 32);
      bs[r] = ssum;
    }
#pragma unroll
    for (int r = 0; r < 8; ++r) rsum[r] = rsum[r] * sc[r] + bs[r];
#pragma unroll
    for (int t = 0; t < 4; ++t)
#pragma unroll
      for (int r = 0; r < 8; ++r) o[t][r] *= sc[r];

    // wave-private LDS region: only need this wave's DS stores to land
    asm volatile("s_wait_dscnt 0" ::: "memory");

    // re-read P in A-fragment layout (row = rA, K = key offset 0..63)
    v16h pf0 = frag_from_f16(P + rA * 64 + hs);
    v16h pf1 = frag_from_f16(P + rA * 64 + 32 + hs);
#pragma unroll
    for (int t = 0; t < 4; ++t) {
      // B[k][n] from V^T: lane holds column d -> contiguous keys
      v16h vf0 = frag_from_f16(Vp + (size_t)(t * 16 + rA) * T_ + j0 + hs);
      v16h vf1 = frag_from_f16(Vp + (size_t)(t * 16 + rA) * T_ + j0 + 32 + hs);
      o[t] = wmma16(pf0, vf0, o[t]);
      o[t] = wmma16(pf1, vf1, o[t]);
    }
  }

  // ---- finalize: divide by row sums, emit f16 row-major [B*T][C] ----
#pragma unroll
  for (int r = 0; r < 8; ++r) rsum[r] = 1.0f / rsum[r];
#pragma unroll
  for (int t = 0; t < 4; ++t) {
#pragma unroll
    for (int r = 0; r < 8; ++r) {
      const int tok = q0 + hs + r;
      const int d = t * 16 + rA;
      Out[(size_t)(b * T_ + tok) * C_ + h * D_ + d] =
          (_Float16)(o[t][r] * rsum[r]);
    }
  }
}

// ---------------------------------------------------------------------------
// Host-side launch.  Inputs: x, mask(unused; causality computed analytically),
// Wq,bq, Wk,bk, Wv,bv, Wp,bp.  Workspace: f16 Q,K,V^T,attn-out (~50.3 MB).
// ---------------------------------------------------------------------------
extern "C" void kernel_launch(void* const* d_in, const int* in_sizes, int n_in,
                              void* d_out, int out_size, void* d_ws,
                              size_t ws_size, hipStream_t stream) {
  (void)in_sizes; (void)n_in; (void)out_size; (void)ws_size;
  const float* x  = (const float*)d_in[0];
  const float* Wq = (const float*)d_in[2];
  const float* bq = (const float*)d_in[3];
  const float* Wk = (const float*)d_in[4];
  const float* bk = (const float*)d_in[5];
  const float* Wv = (const float*)d_in[6];
  const float* bv = (const float*)d_in[7];
  const float* Wp = (const float*)d_in[8];
  const float* bp = (const float*)d_in[9];

  const size_t QKV = (size_t)B_ * H_ * T_ * D_;  // 6,291,456 elements
  _Float16* q  = (_Float16*)d_ws;
  _Float16* k  = q + QKV;
  _Float16* vt = k + QKV;
  _Float16* ao = vt + QKV;

  const dim3 blk(128);
  const dim3 gg(M_ / 128, C_ / 64);                 // 64 x 12 workgroups

  proj_gemm<0, false><<<gg, blk, 0, stream>>>(x, Wq, bq, q, 0.125f);  // Q*1/8
  proj_gemm<0, false><<<gg, blk, 0, stream>>>(x, Wk, bk, k, 1.0f);
  proj_gemm<1, false><<<gg, blk, 0, stream>>>(x, Wv, bv, vt, 1.0f);   // V^T

  flash_attn<<<dim3(T_ / 64, B_ * H_), blk, 0, stream>>>(q, k, vt, ao);

  proj_gemm<2, true><<<gg, blk, 0, stream>>>(ao, Wp, bp, d_out, 1.0f);
}
